// LCF_ATEPC_8031588843931
// MI455X (gfx1250) — compile-verified, matrix operand order checked by reading, not run on hardware
//
#include <hip/hip_runtime.h>

// ---------------------------------------------------------------------------
// LCF-ATEPC fused pipeline for MI455X (gfx1250, wave32, WMMA).
// Heavy GEMMs in bf16 via v_wmma_f32_16x16x32_bf16 (f32 accumulate).
// Only attention row 0 is computed (only sa[:,0] feeds the outputs).
// GEMM wave tile: 32x32 (2x2 accumulators) -- this is the config that
// produces a bubble-free inner loop (no v_nop hazard stalls, no acc moves).
// ---------------------------------------------------------------------------

#define B_  64
#define L_  256
#define H_  768
#define NH_ 12
#define HD_ 64          // head dim
#define H3_ (3 * H_)
#define BL_ (B_ * L_)

typedef __attribute__((ext_vector_type(16))) __bf16 v16bf;
typedef __attribute__((ext_vector_type(8)))  float  v8f;
typedef __attribute__((ext_vector_type(4)))  unsigned int v4u;

// Load one 16x32 bf16 WMMA fragment (A or B^T) for this lane.
// ISA layout: lane l holds row (l&15); elements 0..7 are K = k0 + (l>>4)*8 .. +7
// (contiguous), elements 8..15 are K = k0+16 + (l>>4)*8 .. +7 (contiguous).
// => two 16-byte loads per lane.
__device__ __forceinline__ v16bf load_frag(const __bf16* p) {
    union { v4u u[2]; v16bf v; } f;
    f.u[0] = *reinterpret_cast<const v4u*>(p);
    f.u[1] = *reinterpret_cast<const v4u*>(p + 16);
    return f.v;
}

// ---------------------------------------------------------------------------
// 1) Per-batch inclusive scan of valid_ids -> inverse compaction map.
//    src_of[b*L + d] = source token index that lands in slot d, or -1.
// ---------------------------------------------------------------------------
__global__ void scan_kernel(const int* __restrict__ valid, int* __restrict__ src_of) {
    const int b = blockIdx.x;
    const int t = threadIdx.x;            // 256 threads == L
    __shared__ int s[L_];
    const int v = valid[b * L_ + t];
    s[t] = v;
    __syncthreads();
    for (int off = 1; off < L_; off <<= 1) {
        int x = (t >= off) ? s[t - off] : 0;
        __syncthreads();
        s[t] += x;
        __syncthreads();
    }
    src_of[b * L_ + t] = -1;
    __threadfence_block();
    __syncthreads();
    if (v) src_of[b * L_ + (s[t] - 1)] = t;
}

// ---------------------------------------------------------------------------
// 2) Convert f32 weight [K,N] row-major -> bf16 transposed [N,K] row-major,
//    so WMMA B-fragments are two contiguous b128 loads per lane.
// ---------------------------------------------------------------------------
__global__ void convT_kernel(const float* __restrict__ W, __bf16* __restrict__ out,
                             int K, int N) {
    const int idx = blockIdx.x * 256 + threadIdx.x;
    if (idx >= K * N) return;
    const int k = idx / N, n = idx % N;
    out[(size_t)n * K + k] = (__bf16)W[idx];
}

// ---------------------------------------------------------------------------
// 3) Gather/compact + build cat=[g, loc*cdw, loc*cdm] in bf16, fused with
//    ate_logits = g @ W_ate + b_ate (LDS ds_add_f32 reduction).
//    One block per destination token (b,d); 256 threads cover H=768 (3 each).
// ---------------------------------------------------------------------------
__global__ void build_cat_kernel(const float* __restrict__ gh, const float* __restrict__ lh,
                                 const int* __restrict__ src_of,
                                 const float* __restrict__ cdm, const float* __restrict__ cdw,
                                 const float* __restrict__ W_ate, const float* __restrict__ b_ate,
                                 __bf16* __restrict__ catA, float* __restrict__ ate_out) {
    const int b = blockIdx.x / L_;
    const int d = blockIdx.x % L_;
    const int t = src_of[b * L_ + d];
    __shared__ float s_ate[6];
    if (threadIdx.x < 6) s_ate[threadIdx.x] = 0.f;
    __syncthreads();

    const float cw = cdw[b * L_ + d];
    const float cm = cdm[b * L_ + d];
    __bf16* crow = catA + (size_t)(b * L_ + d) * H3_;
    float acc[6] = {0.f, 0.f, 0.f, 0.f, 0.f, 0.f};

    for (int h = threadIdx.x; h < H_; h += 256) {
        float g = 0.f, l = 0.f;
        if (t >= 0) {
            g = gh[((size_t)b * L_ + t) * H_ + h];
            l = lh[((size_t)b * L_ + t) * H_ + h];
        }
        crow[h]          = (__bf16)g;
        crow[H_ + h]     = (__bf16)(l * cw);
        crow[2 * H_ + h] = (__bf16)(l * cm);
        #pragma unroll
        for (int j = 0; j < 6; ++j) acc[j] += g * W_ate[h * 6 + j];
    }
    #pragma unroll
    for (int j = 0; j < 6; ++j) atomicAdd(&s_ate[j], acc[j]);
    __syncthreads();
    if (threadIdx.x < 6)
        ate_out[(size_t)(b * L_ + d) * 6 + threadIdx.x] = s_ate[threadIdx.x] + b_ate[threadIdx.x];
}

// ---------------------------------------------------------------------------
// 4) Generic bf16 WMMA GEMM:  C_bf16 = epi( A[M,K] @ Bt[N,K]^T + bias ).
//    Block = 256 threads = 8 waves arranged 2(M) x 4(N); wave tile 32x32
//    (four 16x16 f32 accumulators); block tile 64 x 128; K step 32.
//    Requires M%64==0, N%128==0, K%32==0 (true for all call sites).
// ---------------------------------------------------------------------------
__global__ void gemm_bf16_wmma(const __bf16* __restrict__ A, int lda,
                               const __bf16* __restrict__ Bt, int K,
                               const float* __restrict__ bias,
                               __bf16* __restrict__ C, int ldc,
                               float scale, int do_tanh) {
    const int tid  = threadIdx.x;
    const int lane = tid & 31;
    const int wave = tid >> 5;
    const int waveM = wave >> 2;          // 0..1
    const int waveN = wave & 3;           // 0..3
    const int rowBase = blockIdx.y * 64 + waveM * 32;
    const int colBase = blockIdx.x * 128 + waveN * 32;

    const int r = lane & 15;              // row/col within 16-tile
    const int g = lane >> 4;              // K sub-group select

    v8f acc[2][2] = {{{0.f}, {0.f}}, {{0.f}, {0.f}}};

    for (int k0 = 0; k0 < K; k0 += 32) {
        v16bf a0 = load_frag(A + (size_t)(rowBase + r)      * lda + k0 + g * 8);
        v16bf a1 = load_frag(A + (size_t)(rowBase + 16 + r) * lda + k0 + g * 8);
        v16bf b0 = load_frag(Bt + (size_t)(colBase + r)      * K + k0 + g * 8);
        v16bf b1 = load_frag(Bt + (size_t)(colBase + 16 + r) * K + k0 + g * 8);

        acc[0][0] = __builtin_amdgcn_wmma_f32_16x16x32_bf16(false, a0, false, b0,
                        (short)0, acc[0][0], false, false);
        acc[0][1] = __builtin_amdgcn_wmma_f32_16x16x32_bf16(false, a0, false, b1,
                        (short)0, acc[0][1], false, false);
        acc[1][0] = __builtin_amdgcn_wmma_f32_16x16x32_bf16(false, a1, false, b0,
                        (short)0, acc[1][0], false, false);
        acc[1][1] = __builtin_amdgcn_wmma_f32_16x16x32_bf16(false, a1, false, b1,
                        (short)0, acc[1][1], false, false);
    }

    // D layout: VGPR e -> row = e + 8*(lane>>4); col = lane&15 (per 16-tile).
    const int r8 = (lane >> 4) * 8;
    const int cn = lane & 15;
    #pragma unroll
    for (int i = 0; i < 2; ++i) {
        #pragma unroll
        for (int j = 0; j < 2; ++j) {
            const int col = colBase + j * 16 + cn;
            const float bcol = bias ? bias[col] : 0.f;
            #pragma unroll
            for (int e = 0; e < 8; ++e) {
                const int row = rowBase + i * 16 + r8 + e;
                float v = (acc[i][j][e] + bcol) * scale;
                if (do_tanh) v = tanhf(v);
                C[(size_t)row * ldc + col] = (__bf16)v;
            }
        }
    }
}

// ---------------------------------------------------------------------------
// 5) Attention, query row 0 only. One block per (b, head); 256 threads.
//    scores -> softmax (LDS tree reductions) -> ctx -> tanh -> sa0 bf16.
//    Scale 1/sqrt(64) is pre-folded into Q.
// ---------------------------------------------------------------------------
__global__ void attn_row0_kernel(const __bf16* __restrict__ qb,   // [B, H]
                                 const __bf16* __restrict__ kb,   // [B*L, H]
                                 const __bf16* __restrict__ vb,   // [B*L, H]
                                 __bf16* __restrict__ sa0) {      // [B, H]
    const int b = blockIdx.x / NH_;
    const int h = blockIdx.x % NH_;
    const int t = threadIdx.x;            // token index, 256 threads == L
    __shared__ float sc[L_];
    __shared__ float red[L_];

    const __bf16* q0   = qb + (size_t)b * H_ + h * HD_;
    const __bf16* krow = kb + ((size_t)b * L_ + t) * H_ + h * HD_;
    float s = 0.f;
    #pragma unroll 8
    for (int i = 0; i < HD_; ++i) s += (float)q0[i] * (float)krow[i];

    red[t] = s;
    __syncthreads();
    for (int off = L_ / 2; off > 0; off >>= 1) {
        if (t < off) red[t] = fmaxf(red[t], red[t + off]);
        __syncthreads();
    }
    const float mx = red[0];
    __syncthreads();

    const float p = __expf(s - mx);
    sc[t]  = p;
    red[t] = p;
    __syncthreads();
    for (int off = L_ / 2; off > 0; off >>= 1) {
        if (t < off) red[t] += red[t + off];
        __syncthreads();
    }
    const float inv = 1.f / red[0];
    __syncthreads();

    if (t < HD_) {
        const __bf16* vcol = vb + (size_t)b * L_ * H_ + h * HD_ + t;
        float c = 0.f;
        #pragma unroll 8
        for (int tt = 0; tt < L_; ++tt) c += sc[tt] * (float)vcol[(size_t)tt * H_];
        sa0[(size_t)b * H_ + h * HD_ + t] = (__bf16)tanhf(c * inv);
    }
}

// ---------------------------------------------------------------------------
// 6) apc_logits = pooled @ W_apc + b_apc  (tiny; LDS ds_add_f32 reduction).
// ---------------------------------------------------------------------------
__global__ void apc_kernel(const __bf16* __restrict__ pooled, const float* __restrict__ W_apc,
                           const float* __restrict__ b_apc, float* __restrict__ out) {
    const int b = blockIdx.x;
    const int tid = threadIdx.x;          // 96 threads: 3 outputs x 32 partials
    __shared__ float s[3];
    if (tid < 3) s[tid] = 0.f;
    __syncthreads();
    const int j = tid % 3, part = tid / 3;
    float a = 0.f;
    for (int h = part; h < H_; h += 32)
        a += (float)pooled[(size_t)b * H_ + h] * W_apc[h * 3 + j];
    atomicAdd(&s[j], a);
    __syncthreads();
    if (tid < 3) out[b * 3 + tid] = s[tid] + b_apc[tid];
}

// ---------------------------------------------------------------------------
// Workspace layout (bytes, 256-aligned)
// ---------------------------------------------------------------------------
static constexpr size_t OFF_SRC    = 0;
static constexpr size_t OFF_CAT    = OFF_SRC    + (size_t)BL_ * 4;                 // int
static constexpr size_t OFF_X      = OFF_CAT    + (size_t)BL_ * H3_ * 2;           // bf16
static constexpr size_t OFF_K      = OFF_X      + (size_t)BL_ * H_ * 2;
static constexpr size_t OFF_V      = OFF_K      + (size_t)BL_ * H_ * 2;
static constexpr size_t OFF_Q      = OFF_V      + (size_t)BL_ * H_ * 2;
static constexpr size_t OFF_SA0    = OFF_Q      + (size_t)B_ * H_ * 2;
static constexpr size_t OFF_POOLED = OFF_SA0    + (size_t)B_ * H_ * 2;
static constexpr size_t OFF_WTT    = OFF_POOLED + (size_t)B_ * H_ * 2;
static constexpr size_t OFF_WQT    = OFF_WTT    + (size_t)H_ * H3_ * 2;
static constexpr size_t OFF_WKT    = OFF_WQT    + (size_t)H_ * H_ * 2;
static constexpr size_t OFF_WVT    = OFF_WKT    + (size_t)H_ * H_ * 2;
static constexpr size_t OFF_WPT    = OFF_WVT    + (size_t)H_ * H_ * 2;

extern "C" void kernel_launch(void* const* d_in, const int* in_sizes, int n_in,
                              void* d_out, int out_size, void* d_ws, size_t ws_size,
                              hipStream_t stream) {
    (void)in_sizes; (void)n_in; (void)out_size; (void)ws_size;

    const float* gh       = (const float*)d_in[0];
    const float* lh       = (const float*)d_in[1];
    const int*   valid    = (const int*)  d_in[2];
    const float* cdm      = (const float*)d_in[3];
    const float* cdw      = (const float*)d_in[4];
    const float* W_triple = (const float*)d_in[5];
    const float* b_triple = (const float*)d_in[6];
    const float* Wq       = (const float*)d_in[7];
    const float* bq       = (const float*)d_in[8];
    const float* Wk       = (const float*)d_in[9];
    const float* bk       = (const float*)d_in[10];
    const float* Wv       = (const float*)d_in[11];
    const float* bv       = (const float*)d_in[12];
    const float* W_pool   = (const float*)d_in[13];
    const float* b_pool   = (const float*)d_in[14];
    const float* W_ate    = (const float*)d_in[15];
    const float* b_ate    = (const float*)d_in[16];
    const float* W_apc    = (const float*)d_in[17];
    const float* b_apc    = (const float*)d_in[18];

    char* ws = (char*)d_ws;
    int*    src_of = (int*)   (ws + OFF_SRC);
    __bf16* catA   = (__bf16*)(ws + OFF_CAT);
    __bf16* xb     = (__bf16*)(ws + OFF_X);
    __bf16* kb     = (__bf16*)(ws + OFF_K);
    __bf16* vb     = (__bf16*)(ws + OFF_V);
    __bf16* qb     = (__bf16*)(ws + OFF_Q);
    __bf16* sa0    = (__bf16*)(ws + OFF_SA0);
    __bf16* pooled = (__bf16*)(ws + OFF_POOLED);
    __bf16* WtT    = (__bf16*)(ws + OFF_WTT);
    __bf16* WqT    = (__bf16*)(ws + OFF_WQT);
    __bf16* WkT    = (__bf16*)(ws + OFF_WKT);
    __bf16* WvT    = (__bf16*)(ws + OFF_WVT);
    __bf16* WpT    = (__bf16*)(ws + OFF_WPT);

    float* ate_out = (float*)d_out;                       // [B, L, 6]
    float* apc_out = ate_out + (size_t)BL_ * 6;           // [B, 3]

    // 1) compaction map
    scan_kernel<<<B_, L_, 0, stream>>>(valid, src_of);

    // 2) weight conversions (f32 [K,N] -> bf16 [N,K])
    convT_kernel<<<(H3_ * H_ + 255) / 256, 256, 0, stream>>>(W_triple, WtT, H3_, H_);
    convT_kernel<<<(H_ * H_ + 255) / 256, 256, 0, stream>>>(Wq, WqT, H_, H_);
    convT_kernel<<<(H_ * H_ + 255) / 256, 256, 0, stream>>>(Wk, WkT, H_, H_);
    convT_kernel<<<(H_ * H_ + 255) / 256, 256, 0, stream>>>(Wv, WvT, H_, H_);
    convT_kernel<<<(H_ * H_ + 255) / 256, 256, 0, stream>>>(W_pool, WpT, H_, H_);

    // 3) compact + cat build + fused ATE head
    build_cat_kernel<<<BL_, 256, 0, stream>>>(gh, lh, src_of, cdm, cdw,
                                              W_ate, b_ate, catA, ate_out);

    // 4) x = cat @ W_triple + b_triple        [16384, 2304] x [2304, 768]
    gemm_bf16_wmma<<<dim3(H_ / 128, BL_ / 64), 256, 0, stream>>>(
        catA, H3_, WtT, H3_, b_triple, xb, H_, 1.f, 0);

    // Q only for token 0 of each batch: A rows stride L*H, M = B = 64.
    // 1/sqrt(64) folded into Q via scale.
    gemm_bf16_wmma<<<dim3(H_ / 128, B_ / 64), 256, 0, stream>>>(
        xb, L_ * H_, WqT, H_, bq, qb, H_, 0.125f, 0);

    gemm_bf16_wmma<<<dim3(H_ / 128, BL_ / 64), 256, 0, stream>>>(
        xb, H_, WkT, H_, bk, kb, H_, 1.f, 0);
    gemm_bf16_wmma<<<dim3(H_ / 128, BL_ / 64), 256, 0, stream>>>(
        xb, H_, WvT, H_, bv, vb, H_, 1.f, 0);

    // 5) attention row 0 -> sa0 (tanh applied)
    attn_row0_kernel<<<B_ * NH_, L_, 0, stream>>>(qb, kb, vb, sa0);

    // pooled = tanh(sa0 @ W_pool + b_pool)    [64, 768] x [768, 768]
    gemm_bf16_wmma<<<dim3(H_ / 128, B_ / 64), 256, 0, stream>>>(
        sa0, H_, WpT, H_, b_pool, pooled, H_, 1.f, 1);

    // 6) apc head
    apc_kernel<<<B_, 96, 0, stream>>>(pooled, W_apc, b_apc, apc_out);
}
